// WindowAttention_13383118095100
// MI455X (gfx1250) — matrix operand order repaired and checked by reading, not compiled, not run on hardware
//
#include <hip/hip_runtime.h>

typedef __attribute__((ext_vector_type(16))) _Float16 v16h;
typedef __attribute__((ext_vector_type(8)))  _Float16 v8h;
typedef __attribute__((ext_vector_type(8)))  float    v8f;
typedef __attribute__((ext_vector_type(4)))  int      v4i;
typedef __attribute__((address_space(1))) v4i g_v4i;   // global (HIP device) int4
typedef __attribute__((address_space(3))) v4i l_v4i;   // LDS int4

#define DIM  192
#define NTOK 64
#define NH   6
#define HD   32
#define NWIN 64
#define QSCALE 0.17677669529663689f   // 32^-0.5

// dynamic LDS layout (bytes):
//  sS    @      0 : 64x64  f32  scores              16384
//  sBT   @  16384 : 225x6  f32  bias table (+pad)    5408
//  sX    @  21792 : 64x192 f16  staged input        24576
//  sQ    @  46368 : 64x32  f16  Q (pre-scaled)       4096
//  sK    @  50464 : 64x32  f16  K                    4096
//  sVt   @  54560 : 32x64  f16  V transposed         4096
//  sO    @  58656 : 64x192 f16  attention output    24576
//  sP    @  83232 : 64x64  f16  softmax probs        8192
//  sW    @  91424 : 3x32x192 f16 per-head QKV slab  36864  (async-staged)
//  sRmax @ 128288 : 64x4   f32  partial row max      1024
//  sRsum @ 129312 : 64x4   f32  partial row sum      1024
#define SMEM_BYTES 130336
#define SLAB_BYTES 12288   // one matrix's 32x192 f16 slab

#if defined(__has_builtin)
#if __has_builtin(__builtin_amdgcn_global_load_async_to_lds_b128)
#define HAVE_ASYNC_LDS 1
#endif
#endif

static __device__ inline void async_copy_b128(const void* g, void* l) {
#ifdef HAVE_ASYNC_LDS
    __builtin_amdgcn_global_load_async_to_lds_b128((g_v4i*)g, (l_v4i*)l, 0, 0);
#else
    *(v8h*)l = *(const v8h*)g;   // synchronous fallback
#endif
}

static __device__ inline void async_wait0() {
#ifdef HAVE_ASYNC_LDS
#if __has_builtin(__builtin_amdgcn_s_wait_asynccnt)
    __builtin_amdgcn_s_wait_asynccnt(0);
#else
    asm volatile("s_wait_asynccnt 0x0" ::: "memory");
#endif
#endif
}

static __device__ inline v16h cat8(v8h lo, v8h hi) {
    return __builtin_shufflevector(lo, hi, 0,1,2,3,4,5,6,7,8,9,10,11,12,13,14,15);
}

// A (16x32 f16) from LDS f16 row-major [.,ld]  (contiguous k = 16*half + s, matching B)
static __device__ inline v16h loadA_lds(const _Float16* base, int ld,
                                        int m, int k0, int lane) {
    const _Float16* r = base + m * ld + k0 + ((lane & 16) ? 16 : 0);
    return cat8(*(const v8h*)r, *(const v8h*)(r + 8));
}
// B (32x16 f16) from LDS f16 row-major [N][K] (operand = W^T)
static __device__ inline v16h loadB_lds(const _Float16* base, int ld,
                                        int n, int k0, int lane) {
    const _Float16* r = base + n * ld + k0 + ((lane & 16) ? 16 : 0);
    return cat8(*(const v8h*)r, *(const v8h*)(r + 8));
}
// B (32x16 f16) from global f16 row-major [N][K]
static __device__ inline v16h loadB_g16(const _Float16* __restrict__ base, int ld,
                                        int n, int k0, int lane) {
    const _Float16* r = base + n * ld + k0 + ((lane & 16) ? 16 : 0);
    return cat8(*(const v8h*)r, *(const v8h*)(r + 8));
}
// B fallback from global f32 (convert on the fly)
static __device__ inline v16h loadB_g32(const float* __restrict__ base, int ld,
                                        int n, int k0, int lane) {
    const float* r = base + n * ld + k0 + ((lane & 16) ? 16 : 0);
    v16h b;
#pragma unroll
    for (int e = 0; e < 16; ++e) b[e] = (_Float16)r[e];
    return b;
}

// one-shot f32 -> f16 weight conversion into workspace
__global__ void cvt_weights(const float* __restrict__ qw, const float* __restrict__ pw,
                            _Float16* __restrict__ qw16, _Float16* __restrict__ pw16) {
    const int stride = gridDim.x * blockDim.x;
    int i = blockIdx.x * blockDim.x + threadIdx.x;
    for (int idx = i; idx < 3 * DIM * DIM; idx += stride) qw16[idx] = (_Float16)qw[idx];
    for (int idx = i; idx < DIM * DIM; idx += stride)     pw16[idx] = (_Float16)pw[idx];
}

template <bool W16>
__global__ __launch_bounds__(256) void win_attn_fused(
    const float* __restrict__ x,          // (B,64,192)
    const float* __restrict__ mask,       // (64,64,64)
    const float* __restrict__ qkv_w,      // (576,192) f32 (fallback)
    const float* __restrict__ qkv_b,      // (576)
    const float* __restrict__ proj_w,     // (192,192) f32 (fallback)
    const float* __restrict__ proj_b,     // (192)
    const float* __restrict__ bias_table, // (225,6)
    const int*   __restrict__ rel_idx,    // (64,64)
    const _Float16* __restrict__ qw16,    // (576,192) f16 in ws (W16 path)
    const _Float16* __restrict__ pw16,    // (192,192) f16 in ws (W16 path)
    float* __restrict__ out)              // (B,64,192)
{
    extern __shared__ __align__(16) char smem[];
    float*    sS    = (float*)smem;
    float*    sBT   = (float*)(smem + 16384);
    _Float16* sX    = (_Float16*)(smem + 21792);
    _Float16* sQ    = (_Float16*)(smem + 46368);
    _Float16* sK    = (_Float16*)(smem + 50464);
    _Float16* sVt   = (_Float16*)(smem + 54560);
    _Float16* sO    = (_Float16*)(smem + 58656);
    _Float16* sP    = (_Float16*)(smem + 83232);
    _Float16* sW    = (_Float16*)(smem + 91424);
    float*    sRmax = (float*)(smem + 128288);
    float*    sRsum = (float*)(smem + 129312);

    const int b     = blockIdx.x;
    const int tid   = threadIdx.x;
    const int wave  = tid >> 5;
    const int lane  = tid & 31;
    const int lhalf = (lane & 16) ? 8 : 0;  // C/D row offset
    const int lcol  = lane & 15;            // C/D column within tile
    const int w     = b & (NWIN - 1);       // mask window = b % 64

    const float* xb    = x    + (size_t)b * (NTOK * DIM);
    const float* maskw = mask + (size_t)w * (NTOK * NTOK);

    // ---- kick off async staging of head-0 weight slab ----
    if constexpr (W16) {
        const char* base = (const char*)qw16;
        char* dst = (char*)sW;
#pragma unroll
        for (int i = 0; i < 9; ++i) {
            const int j = tid + 256 * i;                 // 0..2303 b128 chunks
            const int matid = j / 768;                   // which of Q/K/V slab
            const int within = j * 16 - matid * SLAB_BYTES;
            async_copy_b128(base + (size_t)matid * 73728 + within,
                            dst + matid * SLAB_BYTES + within);
        }
    }

    // ---- stage x (f32 -> f16) and bias table into LDS, once per block ----
    {
        const float4* xv = (const float4*)xb;  // 3072 float4
        for (int i = tid; i < (NTOK * DIM) / 4; i += 256) {
            float4 v = xv[i];
            _Float16* d = sX + i * 4;
            d[0] = (_Float16)v.x; d[1] = (_Float16)v.y;
            d[2] = (_Float16)v.z; d[3] = (_Float16)v.w;
        }
        for (int i = tid; i < 225 * NH; i += 256) sBT[i] = bias_table[i];
    }

    for (int h = 0; h < NH; ++h) {
        if constexpr (W16) async_wait0();   // own async arrivals done
        __syncthreads();                    // everyone's arrivals visible; x/sBT ready (h==0)

        // ---------- QKV for head h: 3 mats x (4x2) tiles = 24 tiles / 8 waves ----------
#pragma unroll
        for (int it = 0; it < 3; ++it) {
            const int t     = wave + 8 * it;       // 0..23
            const int matid = t >> 3;              // 0=Q 1=K 2=V
            const int rt    = (t & 7) >> 1;        // row tile 0..3
            const int ct    = t & 1;               // col tile 0..1
            const int m0    = rt * 16;
            const int d0    = ct * 16;
            const int wrow  = matid * DIM + h * HD + d0 + lcol;
            v8f acc = {};
#pragma unroll
            for (int kk = 0; kk < 6; ++kk) {
                v16h a = loadA_lds(sX, DIM, m0 + lcol, kk * 32, lane);
                v16h bm;
                if constexpr (W16)
                    bm = loadB_lds(sW + matid * (SLAB_BYTES / 2), DIM,
                                   d0 + lcol, kk * 32, lane);
                else
                    bm = loadB_g32(qkv_w, DIM, wrow, kk * 32, lane);
                acc = __builtin_amdgcn_wmma_f32_16x16x32_f16(false, a, false, bm,
                                                             (short)0, acc, false, false);
            }
            const float bias = qkv_b[wrow];
            if (matid == 0) {
#pragma unroll
                for (int r = 0; r < 8; ++r)
                    sQ[(m0 + lhalf + r) * HD + d0 + lcol] =
                        (_Float16)((acc[r] + bias) * QSCALE);
            } else if (matid == 1) {
#pragma unroll
                for (int r = 0; r < 8; ++r)
                    sK[(m0 + lhalf + r) * HD + d0 + lcol] = (_Float16)(acc[r] + bias);
            } else {  // V stored transposed: sVt[d][m]
#pragma unroll
                for (int r = 0; r < 8; ++r)
                    sVt[(d0 + lcol) * NTOK + (m0 + lhalf + r)] = (_Float16)(acc[r] + bias);
            }
        }
        __syncthreads();   // all waves done reading sW -> safe to restage

        if constexpr (W16) {
            if (h + 1 < NH) {  // overlap next head's weight DMA with attention phases
                const char* base = (const char*)qw16;
                char* dst = (char*)sW;
#pragma unroll
                for (int i = 0; i < 9; ++i) {
                    const int j = tid + 256 * i;
                    const int matid = j / 768;
                    const int within = j * 16 - matid * SLAB_BYTES;
                    async_copy_b128(base + (size_t)matid * 73728 +
                                        (size_t)(h + 1) * SLAB_BYTES + within,
                                    dst + matid * SLAB_BYTES + within);
                }
            } else {           // last head: warm GL2 for projection weights
#pragma unroll
                for (int k2 = 0; k2 < 3; ++k2) {
                    const int line = tid + 256 * k2;   // 576 x 128B lines
                    if (line < 576)
                        __builtin_prefetch((const char*)pw16 + line * 128, 0, 1);
                }
            }
        }

        // ---------- S = Qs @ K^T : 16 tiles / 8 waves ----------
#pragma unroll
        for (int it = 0; it < 2; ++it) {
            const int t  = wave + 8 * it;
            const int m0 = (t >> 2) * 16;
            const int n0 = (t & 3) * 16;
            v16h a  = loadA_lds(sQ, HD, m0 + lcol, 0, lane);
            v16h bm = loadB_lds(sK, HD, n0 + lcol, 0, lane);
            v8f acc = {};
            acc = __builtin_amdgcn_wmma_f32_16x16x32_f16(false, a, false, bm,
                                                         (short)0, acc, false, false);
#pragma unroll
            for (int r = 0; r < 8; ++r)
                sS[(m0 + lhalf + r) * NTOK + n0 + lcol] = acc[r];
        }
        __syncthreads();

        // ---------- softmax (+rel-pos bias, +mask): all 256 threads,
        //            4 threads per row, 16 columns each ----------
        {
            const int m  = tid & 63;
            const int q  = tid >> 6;         // 0..3 quarter of the row
            const int c0 = q * 16;
            const int*   ridx = rel_idx + m * NTOK + c0;
            const float* mrow = maskw   + m * NTOK + c0;
            float* srow = sS + m * NTOK + c0;

            float mx = -1e30f;
#pragma unroll
            for (int c = 0; c < 16; ++c) {
                float s = srow[c] + sBT[ridx[c] * NH + h] + mrow[c];
                srow[c] = s;
                mx = fmaxf(mx, s);
            }
            sRmax[m * 4 + q] = mx;
            __syncthreads();
            mx = fmaxf(fmaxf(sRmax[m * 4 + 0], sRmax[m * 4 + 1]),
                       fmaxf(sRmax[m * 4 + 2], sRmax[m * 4 + 3]));
            float sum = 0.f;
#pragma unroll
            for (int c = 0; c < 16; ++c) {
                float e = __expf(srow[c] - mx);
                srow[c] = e;
                sum += e;
            }
            sRsum[m * 4 + q] = sum;
            __syncthreads();
            sum = (sRsum[m * 4 + 0] + sRsum[m * 4 + 1]) +
                  (sRsum[m * 4 + 2] + sRsum[m * 4 + 3]);
            const float inv = 1.f / sum;
            _Float16* prow = sP + m * NTOK + c0;
#pragma unroll
            for (int c = 0; c < 16; ++c) prow[c] = (_Float16)(srow[c] * inv);
        }
        __syncthreads();

        // ---------- O_h = P @ V : 8 tiles, one per wave ----------
        {
            const int m0 = (wave >> 1) * 16;
            const int d0 = (wave & 1) * 16;
            v8f acc = {};
#pragma unroll
            for (int kk = 0; kk < 2; ++kk) {
                v16h a  = loadA_lds(sP, NTOK, m0 + lcol, kk * 32, lane);
                v16h bm = loadB_lds(sVt, NTOK, d0 + lcol, kk * 32, lane);
                acc = __builtin_amdgcn_wmma_f32_16x16x32_f16(false, a, false, bm,
                                                             (short)0, acc, false, false);
            }
#pragma unroll
            for (int r = 0; r < 8; ++r)
                sO[(m0 + lhalf + r) * DIM + h * HD + d0 + lcol] = (_Float16)acc[r];
        }
        __syncthreads();
    }

    // ---------- final projection: Y = O @ proj_w^T + proj_b : 48 tiles / 8 waves ----------
    float* ob = out + (size_t)b * (NTOK * DIM);
#pragma unroll
    for (int it = 0; it < 6; ++it) {
        const int t  = wave + 8 * it;
        const int m0 = (t / 12) * 16;
        const int n0 = (t % 12) * 16;
        v8f acc = {};
#pragma unroll
        for (int kk = 0; kk < 6; ++kk) {
            v16h a = loadA_lds(sO, DIM, m0 + lcol, kk * 32, lane);
            v16h bm;
            if constexpr (W16) bm = loadB_g16(pw16, DIM, n0 + lcol, kk * 32, lane);
            else               bm = loadB_g32(proj_w, DIM, n0 + lcol, kk * 32, lane);
            acc = __builtin_amdgcn_wmma_f32_16x16x32_f16(false, a, false, bm,
                                                         (short)0, acc, false, false);
        }
        const float bias = proj_b[n0 + lcol];
#pragma unroll
        for (int r = 0; r < 8; ++r)
            ob[(m0 + lhalf + r) * DIM + n0 + lcol] = acc[r] + bias;
    }
}

extern "C" void kernel_launch(void* const* d_in, const int* in_sizes, int n_in,
                              void* d_out, int out_size, void* d_ws, size_t ws_size,
                              hipStream_t stream) {
    const float* x          = (const float*)d_in[0];
    const float* mask       = (const float*)d_in[1];
    const float* qkv_w      = (const float*)d_in[2];
    const float* qkv_b      = (const float*)d_in[3];
    const float* proj_w     = (const float*)d_in[4];
    const float* proj_b     = (const float*)d_in[5];
    const float* bias_table = (const float*)d_in[6];
    const int*   rel_idx    = (const int*)d_in[7];
    float* out = (float*)d_out;

    const int B = in_sizes[0] / (NTOK * DIM);
    const size_t need = (size_t)(3 * DIM * DIM + DIM * DIM) * sizeof(_Float16);

    if (ws_size >= need) {
        _Float16* qw16 = (_Float16*)d_ws;
        _Float16* pw16 = qw16 + 3 * DIM * DIM;
        hipLaunchKernelGGL(cvt_weights, dim3(144), dim3(256), 0, stream,
                           qkv_w, proj_w, qw16, pw16);
        (void)hipFuncSetAttribute((const void*)win_attn_fused<true>,
                                  hipFuncAttributeMaxDynamicSharedMemorySize, SMEM_BYTES);
        hipLaunchKernelGGL(HIP_KERNEL_NAME(win_attn_fused<true>),
                           dim3(B), dim3(256), SMEM_BYTES, stream,
                           x, mask, qkv_w, qkv_b, proj_w, proj_b, bias_table, rel_idx,
                           qw16, pw16, out);
    } else {
        (void)hipFuncSetAttribute((const void*)win_attn_fused<false>,
                                  hipFuncAttributeMaxDynamicSharedMemorySize, SMEM_BYTES);
        hipLaunchKernelGGL(HIP_KERNEL_NAME(win_attn_fused<false>),
                           dim3(B), dim3(256), SMEM_BYTES, stream,
                           x, mask, qkv_w, qkv_b, proj_w, proj_b, bias_table, rel_idx,
                           (const _Float16*)nullptr, (const _Float16*)nullptr, out);
    }
}
